// MultiheadAttention_415
// MI455X (gfx1250) — matrix operand order repaired and checked by reading, not compiled, not run on hardware
//
#include <hip/hip_runtime.h>
#include <hip/hip_bf16.h>

typedef __bf16 bf16_t;
typedef __attribute__((ext_vector_type(16))) __bf16 v16bf;
typedef __attribute__((ext_vector_type(8)))  __bf16 v8bf;
typedef __attribute__((ext_vector_type(8)))  float  v8f;
typedef __attribute__((__vector_size__(4 * sizeof(int)))) int v4i;

#define BATCH  4
#define SEQ    2048
#define DMODEL 1024
#define HEADS  16
#define DK     64

#define ASPACE(n) __attribute__((address_space(n)))

#if __has_builtin(__builtin_amdgcn_global_load_async_to_lds_b128)
#define HAS_ASYNC 1
#else
#define HAS_ASYNC 0
#endif

// ---------------------------------------------------------------------------
// helpers
// ---------------------------------------------------------------------------
// 16-byte global -> LDS copy; async DMA (ASYNCcnt) when available.
__device__ __forceinline__ void copy16_to_lds(const bf16_t* g, bf16_t* l) {
#if HAS_ASYNC
  ASPACE(1) v4i* gp = (ASPACE(1) v4i*)(uintptr_t)g;
  ASPACE(3) v4i* lp = (ASPACE(3) v4i*)(uint32_t)(uintptr_t)l;
  __builtin_amdgcn_global_load_async_to_lds_b128(gp, lp, 0, 0);
#else
  *(v8bf*)l = *(const v8bf*)g;
#endif
}

// Wait until at most N of this thread's async copies remain in flight.
template <int N>
__device__ __forceinline__ void staging_wait_n() {
#if HAS_ASYNC
#if __has_builtin(__builtin_amdgcn_s_wait_asynccnt)
  __builtin_amdgcn_s_wait_asynccnt(N);
#else
  asm volatile("s_wait_asynccnt %0" ::"i"(N) : "memory");
#endif
#endif
}

__device__ __forceinline__ v16bf make_frag(v8bf lo, v8bf hi) {
  v16bf r;
#pragma unroll
  for (int i = 0; i < 8; ++i) { r[i] = lo[i]; r[i + 8] = hi[i]; }
  return r;
}

__device__ __forceinline__ v8f wmma_bf16(v16bf a, v16bf b, v8f c) {
  // D(f32,16x16) = A(bf16,16x32) * B(bf16,32x16) + C
  return __builtin_amdgcn_wmma_f32_16x16x32_bf16(false, a, false, b,
                                                 (short)0, c, false, false);
}

__device__ __forceinline__ float half_max(float v) {
#pragma unroll
  for (int m = 1; m < 16; m <<= 1) v = fmaxf(v, __shfl_xor(v, m, 32));
  return v;
}
__device__ __forceinline__ float half_sum(float v) {
#pragma unroll
  for (int m = 1; m < 16; m <<= 1) v += __shfl_xor(v, m, 32);
  return v;
}

// ---------------------------------------------------------------------------
// Kernel 0a: convert q/k/v fp32 -> bf16 (one-shot, bandwidth-trivial)
// grid = (BSD/2048, 3), block = 256
// ---------------------------------------------------------------------------
__global__ __launch_bounds__(256) void cvt_qkv_kernel(
    const float* __restrict__ q, const float* __restrict__ k,
    const float* __restrict__ v, bf16_t* __restrict__ xb) {
  const int z = blockIdx.y;
  const float* src = (z == 0) ? q : (z == 1) ? k : v;
  bf16_t* dst = xb + (size_t)z * BATCH * SEQ * DMODEL;
  size_t i = ((size_t)blockIdx.x * 256 + threadIdx.x) * 8;
#pragma unroll
  for (int e = 0; e < 8; ++e) dst[i + e] = (bf16_t)src[i + e];
}

// ---------------------------------------------------------------------------
// Kernel 0b: convert + transpose weights fp32[k][n] -> bf16[n][k]
// grid = (16,16,4), block = 256, tiled 64x64 via LDS (padded, conflict-free)
// ---------------------------------------------------------------------------
__global__ __launch_bounds__(256) void wtrans_kernel(
    const float* __restrict__ wq, const float* __restrict__ wk,
    const float* __restrict__ wv, const float* __restrict__ wo,
    bf16_t* __restrict__ wT) {
  const int z = blockIdx.z;
  const float* W = (z == 0) ? wq : (z == 1) ? wk : (z == 2) ? wv : wo;
  bf16_t* O = wT + (size_t)z * DMODEL * DMODEL;
  __shared__ bf16_t T[64][65];
  const int tid = threadIdx.x;
  const int kb = blockIdx.x * 64, nb = blockIdx.y * 64;
#pragma unroll
  for (int e = 0; e < 16; ++e) {
    int idx = e * 256 + tid;
    int r = idx >> 6, c = idx & 63;  // r = k row, c = n col (coalesced read)
    T[c][r] = (bf16_t)W[(size_t)(kb + r) * DMODEL + nb + c];
  }
  __syncthreads();
#pragma unroll
  for (int e = 0; e < 16; ++e) {
    int idx = e * 256 + tid;
    int r = idx >> 6, c = idx & 63;  // r = n row, c = k col (coalesced write)
    O[(size_t)(nb + r) * DMODEL + kb + c] = T[r][c];
  }
}

// ---------------------------------------------------------------------------
// Kernel 1: input projections  hX = Xbf16 @ Wt, double-buffered async tiles.
// q gets the 1/sqrt(dk) scale folded in; v is written transposed [b,h,dk,s].
// grid = (M/128, N/64, 3), block = 256 (8 waves)
// ---------------------------------------------------------------------------
__global__ __launch_bounds__(256) void proj_gemm_kernel(
    const bf16_t* __restrict__ xb, const bf16_t* __restrict__ wT,
    bf16_t* __restrict__ hq, bf16_t* __restrict__ hk,
    bf16_t* __restrict__ hv) {
  const int z = blockIdx.z;
  const bf16_t* X = xb + (size_t)z * BATCH * SEQ * DMODEL;
  const bf16_t* W = wT + (size_t)z * DMODEL * DMODEL;  // [n][k] bf16
  bf16_t* O = (z == 0) ? hq : (z == 1) ? hk : hv;
  const float scl = (z == 0) ? 0.125f : 1.0f;  // fold softmax scale into Q

  __shared__ __align__(16) bf16_t As[2][128][32];  // 16 KB ping-pong
  __shared__ __align__(16) bf16_t Bs[2][64][32];   //  8 KB ping-pong

  const int tid  = threadIdx.x;
  const int lane = tid & 31;
  const int wid  = tid >> 5;
  const int l16  = lane & 15;
  const int hi   = lane >> 4;
  const int wm   = (wid & 3) * 32;
  const int wn   = (wid >> 2) * 32;
  const int mBase = blockIdx.x * 128;
  const int nBase = blockIdx.y * 64;
  const int srow = tid >> 2, scol = (tid & 3) * 8;  // staging chunk coords

  auto stage = [&](int buf, int k0) {
    copy16_to_lds(X + (size_t)(mBase + srow) * DMODEL + k0 + scol,
                  &As[buf][srow][scol]);
    copy16_to_lds(X + (size_t)(mBase + 64 + srow) * DMODEL + k0 + scol,
                  &As[buf][64 + srow][scol]);
    copy16_to_lds(W + (size_t)(nBase + srow) * DMODEL + k0 + scol,
                  &Bs[buf][srow][scol]);
  };

  const v8f vzero = {0.f, 0.f, 0.f, 0.f, 0.f, 0.f, 0.f, 0.f};
  v8f acc[2][2];
#pragma unroll
  for (int i = 0; i < 2; ++i)
#pragma unroll
    for (int j = 0; j < 2; ++j) acc[i][j] = vzero;

  constexpr int NIT = DMODEL / 32;
  stage(0, 0);
  for (int it = 0; it < NIT; ++it) {
    const int buf = it & 1;
    if (it + 1 < NIT) {
      stage(buf ^ 1, (it + 1) * 32);  // prefetch next tile via DMA
      staging_wait_n<3>();            // current tile's 3 copies retired
    } else {
      staging_wait_n<0>();
    }
    __syncthreads();

    v16bf af[2], bfr[2];
#pragma unroll
    for (int rg = 0; rg < 2; ++rg) {
      int row = wm + rg * 16 + l16;
      af[rg] = make_frag(*(const v8bf*)&As[buf][row][hi * 8],
                         *(const v8bf*)&As[buf][row][hi * 8 + 16]);
    }
#pragma unroll
    for (int cg = 0; cg < 2; ++cg) {
      int nn = wn + cg * 16 + l16;
      bfr[cg] = make_frag(*(const v8bf*)&Bs[buf][nn][hi * 8],
                          *(const v8bf*)&Bs[buf][nn][hi * 8 + 16]);
    }
#pragma unroll
    for (int rg = 0; rg < 2; ++rg)
#pragma unroll
      for (int cg = 0; cg < 2; ++cg)
        acc[rg][cg] = wmma_bf16(af[rg], bfr[cg], acc[rg][cg]);
    __syncthreads();
  }

#pragma unroll
  for (int rg = 0; rg < 2; ++rg)
#pragma unroll
    for (int cg = 0; cg < 2; ++cg) {
      int ncol = nBase + wn + cg * 16 + l16;
      int h = ncol >> 6, d = ncol & 63;
#pragma unroll
      for (int r = 0; r < 8; ++r) {
        int mrow = mBase + wm + rg * 16 + r + 8 * hi;
        int bb = mrow >> 11, ss = mrow & 2047;
        size_t bh = (size_t)bb * HEADS + h;
        float val = acc[rg][cg][r] * scl;
        if (z == 2)
          O[(bh * DK + d) * SEQ + ss] = (bf16_t)val;  // V transposed
        else
          O[(bh * SEQ + ss) * DK + d] = (bf16_t)val;
      }
    }
}

// ---------------------------------------------------------------------------
// Kernel 2: flash attention per (b,h).  grid = (S/128, B*H), block = 256.
// Double-buffered async K / pre-transposed V tiles.
// ---------------------------------------------------------------------------
__global__ __launch_bounds__(256) void flash_attn_kernel(
    const bf16_t* __restrict__ HQ, const bf16_t* __restrict__ HK,
    const bf16_t* __restrict__ HV, bf16_t* __restrict__ AO) {
  const int bh = blockIdx.y;
  const int b = bh >> 4, h = bh & 15;
  const int tid  = threadIdx.x;
  const int lane = tid & 31;
  const int wid  = tid >> 5;
  const int l16  = lane & 15;
  const int hi   = lane >> 4;
  const int q0   = blockIdx.x * 128 + wid * 16;

  const bf16_t* Qp = HQ + (size_t)bh * SEQ * DK;  // [s][dk]
  const bf16_t* Kp = HK + (size_t)bh * SEQ * DK;  // [s][dk]
  const bf16_t* Vp = HV + (size_t)bh * DK * SEQ;  // [dk][s] (pre-transposed)

  __shared__ __align__(16) bf16_t Ks[2][32][64];   // 8 KB ping-pong
  __shared__ __align__(16) bf16_t Vt[2][64][32];   // 8 KB ping-pong
  __shared__ __align__(16) bf16_t Ps[8][16][32];   // 8 KB per-wave P scratch

  const int krow = tid >> 3, kcol = (tid & 7) * 8;
  const int vrow = tid >> 2, vcol = (tid & 3) * 8;
  auto stage_kv = [&](int buf, int j) {
    copy16_to_lds(Kp + (size_t)(j + krow) * DK + kcol, &Ks[buf][krow][kcol]);
    copy16_to_lds(Vp + (size_t)vrow * SEQ + j + vcol, &Vt[buf][vrow][vcol]);
  };

  // Q fragments (A layout), scores pre-scaled via projection epilogue
  const bf16_t* qrow = Qp + (size_t)(q0 + l16) * DK;
  v16bf qf[2];
#pragma unroll
  for (int f = 0; f < 2; ++f)
    qf[f] = make_frag(*(const v8bf*)(qrow + f * 32 + hi * 8),
                      *(const v8bf*)(qrow + f * 32 + hi * 8 + 16));

  const v8f vzero = {0.f, 0.f, 0.f, 0.f, 0.f, 0.f, 0.f, 0.f};
  v8f o[4];
#pragma unroll
  for (int t = 0; t < 4; ++t) o[t] = vzero;
  float m_i[8], l_i[8];
#pragma unroll
  for (int r = 0; r < 8; ++r) { m_i[r] = -1e30f; l_i[r] = 0.f; }

  constexpr int NIT = SEQ / 32;
  stage_kv(0, 0);
  for (int it = 0; it < NIT; ++it) {
    const int buf = it & 1;
    if (it + 1 < NIT) {
      stage_kv(buf ^ 1, (it + 1) * 32);  // DMA next key block during compute
      staging_wait_n<2>();
    } else {
      staging_wait_n<0>();
    }
    __syncthreads();

    // S = Q * K^T : two 16-col key tiles, two K-dim fragments each
    v8f s0 = vzero, s1 = vzero;
#pragma unroll
    for (int f = 0; f < 2; ++f) {
      v16bf k0f =
          make_frag(*(const v8bf*)&Ks[buf][l16][f * 32 + hi * 8],
                    *(const v8bf*)&Ks[buf][l16][f * 32 + hi * 8 + 16]);
      v16bf k1f =
          make_frag(*(const v8bf*)&Ks[buf][16 + l16][f * 32 + hi * 8],
                    *(const v8bf*)&Ks[buf][16 + l16][f * 32 + hi * 8 + 16]);
      s0 = wmma_bf16(qf[f], k0f, s0);
      s1 = wmma_bf16(qf[f], k1f, s1);
    }

    // online softmax (rows live in half-waves; 8 rows per lane)
    float alpha[8];
#pragma unroll
    for (int r = 0; r < 8; ++r) {
      float x0 = s0[r], x1 = s1[r];
      float rm = half_max(fmaxf(x0, x1));
      float mn = fmaxf(m_i[r], rm);
      float a  = __expf(m_i[r] - mn);
      float p0 = __expf(x0 - mn);
      float p1 = __expf(x1 - mn);
      l_i[r] = l_i[r] * a + half_sum(p0 + p1);
      m_i[r] = mn;
      alpha[r] = a;
      Ps[wid][r + 8 * hi][l16]      = (bf16_t)p0;
      Ps[wid][r + 8 * hi][16 + l16] = (bf16_t)p1;
    }
    // intra-wave LDS RAW: make P stores visible before A-layout reload
    asm volatile("s_wait_dscnt 0" ::: "memory");

    v16bf pf = make_frag(*(const v8bf*)&Ps[wid][l16][hi * 8],
                         *(const v8bf*)&Ps[wid][l16][hi * 8 + 16]);

    // O = diag(alpha)*O + P @ V
#pragma unroll
    for (int t = 0; t < 4; ++t) {
#pragma unroll
      for (int r = 0; r < 8; ++r) o[t][r] *= alpha[r];
      v16bf vf =
          make_frag(*(const v8bf*)&Vt[buf][t * 16 + l16][hi * 8],
                    *(const v8bf*)&Vt[buf][t * 16 + l16][hi * 8 + 16]);
      o[t] = wmma_bf16(pf, vf, o[t]);
    }
    __syncthreads();
  }

  // normalize and store concat-head [b, s, h*64+d]
#pragma unroll
  for (int r = 0; r < 8; ++r) {
    float inv = 1.0f / l_i[r];
    int srow = q0 + r + 8 * hi;
    bf16_t* op = AO + ((size_t)b * SEQ + srow) * DMODEL + h * DK;
#pragma unroll
    for (int t = 0; t < 4; ++t) op[t * 16 + l16] = (bf16_t)(o[t][r] * inv);
  }
}

// ---------------------------------------------------------------------------
// Kernel 3: output projection  out(f32) = AO(bf16) @ WoT(bf16, [n][k])
// grid = (M/128, N/64), block = 256, double-buffered async tiles
// ---------------------------------------------------------------------------
__global__ __launch_bounds__(256) void out_gemm_kernel(
    const bf16_t* __restrict__ A, const bf16_t* __restrict__ WoT,
    float* __restrict__ out) {
  __shared__ __align__(16) bf16_t As[2][128][32];
  __shared__ __align__(16) bf16_t Bs[2][64][32];

  const int tid  = threadIdx.x;
  const int lane = tid & 31;
  const int wid  = tid >> 5;
  const int l16  = lane & 15;
  const int hi   = lane >> 4;
  const int wm   = (wid & 3) * 32;
  const int wn   = (wid >> 2) * 32;
  const int mBase = blockIdx.x * 128;
  const int nBase = blockIdx.y * 64;
  const int srow = tid >> 2, scol = (tid & 3) * 8;

  auto stage = [&](int buf, int k0) {
    copy16_to_lds(A + (size_t)(mBase + srow) * DMODEL + k0 + scol,
                  &As[buf][srow][scol]);
    copy16_to_lds(A + (size_t)(mBase + 64 + srow) * DMODEL + k0 + scol,
                  &As[buf][64 + srow][scol]);
    copy16_to_lds(WoT + (size_t)(nBase + srow) * DMODEL + k0 + scol,
                  &Bs[buf][srow][scol]);
  };

  const v8f vzero = {0.f, 0.f, 0.f, 0.f, 0.f, 0.f, 0.f, 0.f};
  v8f acc[2][2];
#pragma unroll
  for (int i = 0; i < 2; ++i)
#pragma unroll
    for (int j = 0; j < 2; ++j) acc[i][j] = vzero;

  constexpr int NIT = DMODEL / 32;
  stage(0, 0);
  for (int it = 0; it < NIT; ++it) {
    const int buf = it & 1;
    if (it + 1 < NIT) {
      stage(buf ^ 1, (it + 1) * 32);
      staging_wait_n<3>();
    } else {
      staging_wait_n<0>();
    }
    __syncthreads();

    v16bf af[2], bfr[2];
#pragma unroll
    for (int rg = 0; rg < 2; ++rg) {
      int row = wm + rg * 16 + l16;
      af[rg] = make_frag(*(const v8bf*)&As[buf][row][hi * 8],
                         *(const v8bf*)&As[buf][row][hi * 8 + 16]);
    }
#pragma unroll
    for (int cg = 0; cg < 2; ++cg) {
      int nn = wn + cg * 16 + l16;
      bfr[cg] = make_frag(*(const v8bf*)&Bs[buf][nn][hi * 8],
                          *(const v8bf*)&Bs[buf][nn][hi * 8 + 16]);
    }
#pragma unroll
    for (int rg = 0; rg < 2; ++rg)
#pragma unroll
      for (int cg = 0; cg < 2; ++cg)
        acc[rg][cg] = wmma_bf16(af[rg], bfr[cg], acc[rg][cg]);
    __syncthreads();
  }

#pragma unroll
  for (int rg = 0; rg < 2; ++rg)
#pragma unroll
    for (int cg = 0; cg < 2; ++cg) {
      int ncol = nBase + wn + cg * 16 + l16;
#pragma unroll
      for (int r = 0; r < 8; ++r) {
        int mrow = mBase + wm + rg * 16 + r + 8 * hi;
        out[(size_t)mrow * DMODEL + ncol] = acc[rg][cg][r];
      }
    }
}

// ---------------------------------------------------------------------------
extern "C" void kernel_launch(void* const* d_in, const int* in_sizes, int n_in,
                              void* d_out, int out_size, void* d_ws,
                              size_t ws_size, hipStream_t stream) {
  (void)in_sizes; (void)n_in; (void)out_size; (void)ws_size;
  const float* q  = (const float*)d_in[0];
  const float* k  = (const float*)d_in[1];
  const float* v  = (const float*)d_in[2];
  const float* wq = (const float*)d_in[3];
  const float* wk = (const float*)d_in[4];
  const float* wv = (const float*)d_in[5];
  const float* wo = (const float*)d_in[6];
  float* out = (float*)d_out;

  const size_t per = (size_t)BATCH * SEQ * DMODEL;  // 8Mi elements
  const size_t wsz = (size_t)DMODEL * DMODEL;       // 1Mi elements
  bf16_t* xb = (bf16_t*)d_ws;       // [3][B*S][D] bf16 inputs
  bf16_t* wT = xb + 3 * per;        // [4][N][K] bf16 transposed weights
  bf16_t* hq = wT + 4 * wsz;        // [B,H,S,DK]  (pre-scaled by 1/8)
  bf16_t* hk = hq + per;            // [B,H,S,DK]
  bf16_t* hv = hk + per;            // [B,H,DK,S]  (transposed)
  bf16_t* ao = hv + per;            // [B,S,D] concat heads

  dim3 blk(256);
  cvt_qkv_kernel<<<dim3(per / 2048, 3), blk, 0, stream>>>(q, k, v, xb);
  wtrans_kernel<<<dim3(16, 16, 4), blk, 0, stream>>>(wq, wk, wv, wo, wT);
  proj_gemm_kernel<<<dim3((BATCH * SEQ) / 128, DMODEL / 64, 3), blk, 0,
                     stream>>>(xb, wT, hq, hk, hv);
  flash_attn_kernel<<<dim3(SEQ / 128, BATCH * HEADS), blk, 0, stream>>>(
      hq, hk, hv, ao);
  out_gemm_kernel<<<dim3((BATCH * SEQ) / 128, DMODEL / 64), blk, 0, stream>>>(
      ao, wT + 3 * wsz, out);
}
